// GCN_31696858644576
// MI455X (gfx1250) — matrix operand order repaired and checked by reading, not compiled, not run on hardware
//
#include <hip/hip_runtime.h>
#include <hip/hip_bf16.h>

// ---------------------------------------------------------------------------
// GCN forward for MI455X (gfx1250, wave32, WMMA).
//   layer: h = X @ W ; agg = sum_e dinv[s]dinv[d] h[s] -> dst ; agg += h*dinv^2 + b
//   out = log_softmax(h2 @ W_out + b_out)
// Dense GEMMs use v_wmma_f32_16x16x32_bf16 (fp32 accum). Edge aggregation is
// the bandwidth-bound part: coalesced float4 gathers + global fp32 atomics.
// ---------------------------------------------------------------------------

typedef __attribute__((ext_vector_type(16))) __bf16 v16bf;
typedef __attribute__((ext_vector_type(8)))  float  v8f;

#define N_NODES 50000
#define N_EDGES 800000
#define N_FEAT  128
#define D_HID   128
#define N_CLASS 40
#define NP_OUT  48   // padded output columns (3 n-tiles of 16)

__device__ __forceinline__ unsigned int f32_to_bf16_bits(float f) {
  unsigned int u = __float_as_uint(f);
  unsigned int r = u + 0x7FFFu + ((u >> 16) & 1u);   // round-to-nearest-even
  return r >> 16;
}

// ---------------------- degree / dinv ---------------------------------------
__global__ void deg_kernel(const int* __restrict__ dst, float* __restrict__ deg, int E) {
  int e = blockIdx.x * blockDim.x + threadIdx.x;
  if (e < E) atomicAdd(&deg[dst[e]], 1.0f);
}

__global__ void dinv_kernel(const float* __restrict__ deg, float* __restrict__ dinv, int n) {
  int i = blockIdx.x * blockDim.x + threadIdx.x;
  if (i < n) dinv[i] = rsqrtf(1.0f + deg[i]);
}

// ---------------------- fp32 -> bf16 ----------------------------------------
__global__ void cvt_bf16_kernel(const float* __restrict__ in,
                                unsigned short* __restrict__ out, int n) {
  int i = blockIdx.x * blockDim.x + threadIdx.x;
  if (i < n) out[i] = (unsigned short)f32_to_bf16_bits(in[i]);
}

// ---------------------- pack W into B-fragment order ------------------------
// Bpack dword index = ((kt*ntiles + nt)*32 + lane)*8 + v
//   lane: n = nt*16 + (lane&15), K half = (lane>>4)*16
//   dword v packs bf16 pair (k = kt*32 + half + 2v, k+1)   [lo = k, hi = k+1]
__global__ void pack_w_kernel(const float* __restrict__ W,   // [K, Nw] row-major
                              unsigned int* __restrict__ Bpack,
                              int K, int Nw, int ntiles) {
  int tid = blockIdx.x * blockDim.x + threadIdx.x;
  int ktiles = K >> 5;
  int total = ktiles * ntiles * 32 * 8;
  if (tid >= total) return;
  int v    = tid & 7;
  int lane = (tid >> 3) & 31;
  int nt   = (tid >> 8) % ntiles;
  int kt   = (tid >> 8) / ntiles;
  int n = nt * 16 + (lane & 15);
  int k = kt * 32 + (lane >> 4) * 16 + 2 * v;
  unsigned int lo = 0, hi = 0;
  if (n < Nw) {
    lo = f32_to_bf16_bits(W[(size_t)k * Nw + n]);
    hi = f32_to_bf16_bits(W[(size_t)(k + 1) * Nw + n]);
  }
  Bpack[tid] = lo | (hi << 16);
}

// ---------------------- WMMA GEMM: C[M,NP] = Abf16[M,128] @ Bpack -----------
// One wave per 16x16 output tile; K = 128 hardcoded (4 unrolled WMMAs).
__global__ void __launch_bounds__(32)
wmma_gemm_bf16(const unsigned short* __restrict__ Abf,   // [M,128] bf16 row-major
               const unsigned int*   __restrict__ Bpack, // packed fragments
               float* __restrict__ C, int NP, int ntiles) {
  const int K = 128;
  int mtile = blockIdx.x;
  int nt    = blockIdx.y;
  int lane  = threadIdx.x;
  int half  = lane >> 4;
  size_t arow = (size_t)(mtile * 16 + (lane & 15)) * K;

  v8f acc = {};
#pragma unroll
  for (int kt = 0; kt < 4; ++kt) {
    // A fragment: two 16B loads per lane (contiguous k = 16*half + s)
    union { uint4 q[2]; v16bf v; } ua;
    const char* abase = (const char*)(Abf + arow + kt * 32) + half * 32;
    ua.q[0] = *(const uint4*)(abase);
    ua.q[1] = *(const uint4*)(abase + 16);
    // B fragment: 32B contiguous per lane from packed buffer
    union { uint4 q[2]; v16bf v; } ub;
    const uint4* bbase =
        (const uint4*)(Bpack + ((size_t)(kt * ntiles + nt) * 32 + lane) * 8);
    ub.q[0] = bbase[0];
    ub.q[1] = bbase[1];
    acc = __builtin_amdgcn_wmma_f32_16x16x32_bf16(
        false, ua.v, false, ub.v, (short)0, acc, false, false);
  }
  // C/D layout: col = lane&15, row = r + 8*(lane>>4)
  int col = nt * 16 + (lane & 15);
  int row0 = mtile * 16 + 8 * half;
#pragma unroll
  for (int r = 0; r < 8; ++r)
    C[(size_t)(row0 + r) * NP + col] = acc[r];
}

// ---------------------- edge message scatter-add ----------------------------
// thread = (edge, float4 chunk); consecutive threads cover one source row.
__global__ void edge_agg_kernel(const float* __restrict__ h,
                                const float* __restrict__ dinv,
                                const int* __restrict__ src,
                                const int* __restrict__ dst,
                                float* __restrict__ agg, int E, int D) {
  int chunks = D >> 2;
  int t = blockIdx.x * blockDim.x + threadIdx.x;
  if (t >= E * chunks) return;
  int e = t / chunks;
  int c = t - e * chunks;
  int s = src[e], d = dst[e];
  float coef = dinv[s] * dinv[d];
  const float4 hv = *(const float4*)(h + (size_t)s * D + c * 4);
  float* p = agg + (size_t)d * D + c * 4;
  atomicAdd(p + 0, hv.x * coef);
  atomicAdd(p + 1, hv.y * coef);
  atomicAdd(p + 2, hv.z * coef);
  atomicAdd(p + 3, hv.w * coef);
}

// ---------------------- self-loop + bias (+relu) ----------------------------
__global__ void finalize_kernel(float* __restrict__ agg,
                                const float* __restrict__ h,
                                const float* __restrict__ dinv,
                                const float* __restrict__ bias,
                                int M, int D, int relu) {
  int i = blockIdx.x * blockDim.x + threadIdx.x;
  if (i >= M * D) return;
  int row = i / D;
  int col = i - row * D;
  float di = dinv[row];
  float v = agg[i] + h[i] * di * di + bias[col];
  agg[i] = relu ? fmaxf(v, 0.0f) : v;
}

// ---------------------- bias + row log-softmax (wave32 per row) -------------
__global__ void __launch_bounds__(32)
logsoftmax_kernel(const float* __restrict__ in,   // [M, NP] (NP=48)
                  const float* __restrict__ bias, // [C]
                  float* __restrict__ out,        // [M, C]
                  int NP, int C) {
  int row = blockIdx.x;
  int lane = threadIdx.x;
  const float* rp = in + (size_t)row * NP;
  int c1 = lane + 32;
  float v0 = (lane < C) ? rp[lane] + bias[lane] : -INFINITY;
  float v1 = (c1 < C) ? rp[c1] + bias[c1] : -INFINITY;
  float m = fmaxf(v0, v1);
#pragma unroll
  for (int off = 16; off; off >>= 1) m = fmaxf(m, __shfl_xor(m, off, 32));
  float s = ((lane < C) ? __expf(v0 - m) : 0.0f) + ((c1 < C) ? __expf(v1 - m) : 0.0f);
#pragma unroll
  for (int off = 16; off; off >>= 1) s += __shfl_xor(s, off, 32);
  float ls = __logf(s) + m;
  if (lane < C) out[(size_t)row * C + lane] = v0 - ls;
  if (c1 < C)  out[(size_t)row * C + c1]  = v1 - ls;
}

// ---------------------------------------------------------------------------
extern "C" void kernel_launch(void* const* d_in, const int* in_sizes, int n_in,
                              void* d_out, int out_size, void* d_ws, size_t ws_size,
                              hipStream_t stream) {
  const float* x     = (const float*)d_in[0];
  const int*   edge  = (const int*)d_in[1];
  const float* W_in  = (const float*)d_in[2];
  const float* b_in  = (const float*)d_in[3];
  const float* W1    = (const float*)d_in[4];
  const float* b1    = (const float*)d_in[5];
  const float* W_out = (const float*)d_in[6];
  const float* b_out = (const float*)d_in[7];
  float* out = (float*)d_out;

  const int M = N_NODES, E = N_EDGES, D = D_HID;
  const int* src = edge;
  const int* dst = edge + E;

  // workspace layout (bytes, 256-aligned blocks)
  char* ws = (char*)d_ws;
  float*          deg   = (float*)(ws + 0);               // 200,000 B
  float*          dinv  = (float*)(ws + 204800);          // 200,000 B
  unsigned short* bfA   = (unsigned short*)(ws + 409600); // 12,800,000 B
  float*          h     = (float*)(ws + 13312000);        // 25,600,000 B
  float*          agg   = (float*)(ws + 38912000);        // 25,600,000 B
  unsigned int*   wpack = (unsigned int*)(ws + 64512000); // 32,768 B

  const int mtiles = M / 16;  // 3125

  // --- degree / dinv ---
  hipMemsetAsync(deg, 0, (size_t)M * sizeof(float), stream);
  deg_kernel<<<(E + 255) / 256, 256, 0, stream>>>(dst, deg, E);
  dinv_kernel<<<(M + 255) / 256, 256, 0, stream>>>(deg, dinv, M);

  // ================= layer 1: gcn_conv(x, W_in, b_in) =================
  cvt_bf16_kernel<<<(M * D + 255) / 256, 256, 0, stream>>>(x, bfA, M * D);
  pack_w_kernel<<<(4 * 8 * 256 + 255) / 256, 256, 0, stream>>>(W_in, wpack, 128, 128, 8);
  wmma_gemm_bf16<<<dim3(mtiles, 8), 32, 0, stream>>>(bfA, wpack, h, 128, 8);
  hipMemsetAsync(agg, 0, (size_t)M * D * sizeof(float), stream);
  edge_agg_kernel<<<(E * (D / 4) + 255) / 256, 256, 0, stream>>>(h, dinv, src, dst, agg, E, D);
  finalize_kernel<<<(M * D + 255) / 256, 256, 0, stream>>>(agg, h, dinv, b_in, M, D, 0);

  // ================= layer 2: relu(gcn_conv(h, W1, b1)) =================
  cvt_bf16_kernel<<<(M * D + 255) / 256, 256, 0, stream>>>(agg, bfA, M * D);
  pack_w_kernel<<<(4 * 8 * 256 + 255) / 256, 256, 0, stream>>>(W1, wpack, 128, 128, 8);
  wmma_gemm_bf16<<<dim3(mtiles, 8), 32, 0, stream>>>(bfA, wpack, h, 128, 8);
  hipMemsetAsync(agg, 0, (size_t)M * D * sizeof(float), stream);
  edge_agg_kernel<<<(E * (D / 4) + 255) / 256, 256, 0, stream>>>(h, dinv, src, dst, agg, E, D);
  finalize_kernel<<<(M * D + 255) / 256, 256, 0, stream>>>(agg, h, dinv, b1, M, D, 1);

  // ================= output: log_softmax(h @ W_out + b_out) =================
  cvt_bf16_kernel<<<(M * D + 255) / 256, 256, 0, stream>>>(agg, bfA, M * D);
  pack_w_kernel<<<(4 * 3 * 256 + 255) / 256, 256, 0, stream>>>(W_out, wpack, 128, N_CLASS, 3);
  wmma_gemm_bf16<<<dim3(mtiles, 3), 32, 0, stream>>>(bfA, wpack, h, NP_OUT, 3);
  logsoftmax_kernel<<<M, 32, 0, stream>>>(h, b_out, out, NP_OUT, N_CLASS);
}
